// FewShotRefine_36086315221371
// MI455X (gfx1250) — compile-verified, matrix operand order checked by reading, not run on hardware
//
#include <hip/hip_runtime.h>
#include <hip/hip_bf16.h>
#include <math.h>

// ---------------------------------------------------------------------------
// FewShotRefine for MI455X (gfx1250, wave32, WMMA + TDM).
//
// Shapes: sprotos (4,2,2048) f32 | qft_high (1,4,2048,48,48) f32
//         un_fts (12,2048,48,48) f32
// Out:    output (4,2,417,417) f32 ++ masks (4,2,48,48) f32
//
// Matmul sweeps use v_wmma_f32_16x16x32_bf16 (f32 accumulate), one wave per
// 16x16 tile.  kmeans runs on the 48x48 grid (bilinear upsample is linear,
// ||p||^2 cancels in the argmin) so the 56.6MB/episode feature block stays
// L2-resident (192MB L2).  The centroid-update GEMM stages its shared A
// matrix through LDS with tensor_load_to_lds (TDM), double-buffered on
// TENSORcnt.
// ---------------------------------------------------------------------------

#define C_DIM    2048
#define HW       2304             // 48*48
#define IMG      ((size_t)C_DIM * HW)
#define KPIX     6912             // 3*2304 kmeans reduction length
#define EPS_F    1e-8f
#define SCALER_F 20.0f

typedef __attribute__((ext_vector_type(16))) __bf16 v16bf;
typedef __attribute__((ext_vector_type(8)))  float  v8f;
typedef __attribute__((ext_vector_type(4)))  unsigned int u32x4;
typedef __attribute__((ext_vector_type(8)))  int    i32x8;
typedef __attribute__((ext_vector_type(4)))  int    i32x4;

// ---------------------------------------------------------------------------
// WMMA K-loop: D[16 protos x 16 pixels] += A(16x2048) * feat(2048 x HW) tile.
// A: lane holds row (lane&15), K-half by lane>=16 -> contiguous, b128 loads.
// B: lane holds pixel column (lane&15), K-half by lane>=16 -> strided gather.
// Optionally accumulates sum(f^2) per pixel split across the two lane halves
// (combined with shfl_xor 16 by the caller).
// ---------------------------------------------------------------------------
template <bool WANT_NORM>
__device__ __forceinline__ void wmma_pdot(const float* __restrict__ A,
                                          const float* __restrict__ feat,
                                          int pix0, v8f& acc, float& nsq) {
  const int lane = threadIdx.x & 31;
  const int half = lane >> 4;
  const int col  = lane & 15;
  const float4* __restrict__ arow4 =
      (const float4*)(A + (size_t)col * C_DIM + half * 16);     // 64B aligned
  const float* __restrict__ fcol = feat + pix0 + col;
  for (int k0 = 0; k0 < C_DIM; k0 += 32) {
    if (k0 + 64 < C_DIM)  // -> global_prefetch_b8, 2 chunks ahead
      __builtin_prefetch(fcol + (size_t)(k0 + 64 + half * 16) * HW, 0, 0);
    v16bf a, b;
#pragma unroll
    for (int j4 = 0; j4 < 4; ++j4) {                            // A: 4x b128
      const float4 av = arow4[(k0 >> 2) + j4];
      a[j4 * 4 + 0] = (__bf16)av.x;
      a[j4 * 4 + 1] = (__bf16)av.y;
      a[j4 * 4 + 2] = (__bf16)av.z;
      a[j4 * 4 + 3] = (__bf16)av.w;
    }
#pragma unroll
    for (int j = 0; j < 16; ++j) {                              // B: strided
      const float v = fcol[(size_t)(k0 + half * 16 + j) * HW];
      if (WANT_NORM) nsq += v * v;
      b[j] = (__bf16)v;
    }
    acc = __builtin_amdgcn_wmma_f32_16x16x32_bf16(false, a, false, b,
                                                  (short)0, acc, false, false);
  }
}

// ---------------------------------------------------------------------------
// TDM: DMA a 16-row x (KCH cols) f32 tile of A (row stride KPIX) into LDS.
// D# packing per CDNA5 ISA (group0: count/lds/global/
// type; group1: data_size=4B, tensor_dim0=KPIX, dim1=16, tile KCHx16,
// dim0_stride=KPIX).  6-arg clang-23 builtin form; trailing groups zero (2D).
// ---------------------------------------------------------------------------
#define KCH 256
__device__ __forceinline__ void tdm_load_tile(const float* gsrc, unsigned lds_byte) {
  const unsigned long long ga = (unsigned long long)(uintptr_t)gsrc;
  u32x4 g0;
  g0[0] = 1u;                                            // count=1, user mode
  g0[1] = lds_byte;                                      // lds_addr
  g0[2] = (unsigned)(ga & 0xffffffffu);                  // global_addr[31:0]
  g0[3] = (unsigned)((ga >> 32) & 0x01ffffffu)           // global_addr[56:32]
        | 0x80000000u;                                   // type=2 (bits 127:126)
  i32x8 g1;
  g1[0] = 0x00020000;                                    // data_size=2 (4B)
  g1[1] = (int)((KPIX & 0xffff) << 16);                  // tensor_dim0 lo16
  g1[2] = (int)(((KPIX >> 16) & 0xffff) | (16 << 16));   // dim0 hi16 | dim1 lo16
  g1[3] = (int)(KCH << 16);                              // dim1 hi16=0 | tile_dim0
  g1[4] = 16;                                            // tile_dim1 | tile_dim2=0
  g1[5] = KPIX;                                          // dim0_stride[31:0]
  g1[6] = 0;                                             // dim0_stride hi | dim1_stride lo
  g1[7] = 0;
  const i32x4 z4 = {0, 0, 0, 0};
  const i32x8 z8 = {0, 0, 0, 0, 0, 0, 0, 0};
  __builtin_amdgcn_tensor_load_to_lds(g0, g1, z4, z4, z8, 0);
}

// ---------------------------------------------------------------------------
// 1. Normalize bg/fg protos into a zero-padded 16x2048 A-matrix.
// ---------------------------------------------------------------------------
__global__ void k_norm_protos(const float* __restrict__ sp, float* __restrict__ out16,
                              int epi) {
  __shared__ float red[256];
  const int row = blockIdx.x, t = threadIdx.x;
  const float* src = (row < 2) ? sp + ((size_t)epi * 2 + row) * C_DIM : nullptr;
  float s = 0.f;
  if (src) for (int c = t; c < C_DIM; c += 256) { float v = src[c]; s += v * v; }
  red[t] = s; __syncthreads();
  for (int st = 128; st > 0; st >>= 1) { if (t < st) red[t] += red[t + st]; __syncthreads(); }
  const float inv = src ? 1.f / fmaxf(sqrtf(red[0]), EPS_F) : 0.f;
  for (int c = t; c < C_DIM; c += 256)
    out16[(size_t)row * C_DIM + c] = src ? src[c] * inv : 0.f;
}

// ---------------------------------------------------------------------------
// 2. Initial cal_dist + softmax + round -> hard masks, per-pixel inv norms.
//    grid (144 tiles, 4 images) x 32 threads (1 wave).
// ---------------------------------------------------------------------------
__global__ void k_init_caldist(const float* __restrict__ qimg,
                               const float* __restrict__ uimgs,
                               const float* __restrict__ protoN2,
                               float* __restrict__ invnorm,
                               float* __restrict__ hard,
                               float* __restrict__ masks_out, int epi) {
  const int tile = blockIdx.x, img = blockIdx.y;
  const float* feat = (img == 0) ? qimg : uimgs + (size_t)(img - 1) * IMG;
  const int pix0 = tile * 16;
  v8f acc = {};
  float nsq = 0.f;
  wmma_pdot<true>(protoN2, feat, pix0, acc, nsq);
  nsq += __shfl_xor(nsq, 16);
  const int lane = threadIdx.x & 31;
  if (lane < 16) {
    const int pix = pix0 + lane;
    const float inv = 1.f / fmaxf(sqrtf(nsq), EPS_F);
    const float sb = SCALER_F * inv * acc[0];   // D row 0 = bg
    const float sf = SCALER_F * inv * acc[1];   // D row 1 = fg
    const float pb = 1.f / (1.f + __expf(sf - sb));
    const float pf = 1.f - pb;
    const float hb = rintf(pb), hf = rintf(pf); // jnp.round = half-to-even
    invnorm[(size_t)img * HW + pix] = inv;
    hard[((size_t)img * 2 + 0) * HW + pix] = hb;
    hard[((size_t)img * 2 + 1) * HW + pix] = hf;
    if (img == 0) {
      masks_out[((size_t)epi * 2 + 0) * HW + pix] = hb;
      masks_out[((size_t)epi * 2 + 1) * HW + pix] = hf;
    }
  }
}

// ---------------------------------------------------------------------------
// 3. Centroid init: cent0 = upsampled_points[linspace idx]; both sides share.
//    Pads rows 5..15 with zeros.  grid (16 rows, 8) x 256.
// ---------------------------------------------------------------------------
__global__ void k_cent_init(const float* __restrict__ uimgs, float* __restrict__ cent) {
  const int ch = blockIdx.y * 256 + threadIdx.x;
  const int row = blockIdx.x;
  float v = 0.f;
  if (row < 5) {
    const int idxs[5] = {0, 6911, 13823, 20735, 27647};
    const int idx = idxs[row];
    const int img = idx / 9216, rem = idx % 9216;
    const int Y = rem / 96, X = rem % 96;
    const float ys = Y * 47.f / 95.f, xs = X * 47.f / 95.f;
    const int y0 = (int)floorf(ys); const int y1 = (y0 + 1 < 47) ? y0 + 1 : 47;
    const int x0 = (int)floorf(xs); const int x1 = (x0 + 1 < 47) ? x0 + 1 : 47;
    const float wy = ys - (float)y0, wx = xs - (float)x0;
    const float* f = uimgs + (size_t)img * IMG + (size_t)ch * HW;
    const float t0 = f[y0 * 48 + x0] * (1.f - wy) + f[y1 * 48 + x0] * wy;
    const float t1 = f[y0 * 48 + x1] * (1.f - wy) + f[y1 * 48 + x1] * wy;
    v = t0 * (1.f - wx) + t1 * wx;
  }
  cent[(size_t)row * C_DIM + ch] = v;                      // side 0 (bg)
  cent[(size_t)(16 + row) * C_DIM + ch] = v;               // side 1 (fg)
}

// ---------------------------------------------------------------------------
// 4. ||c_k||^2 per (side, cluster).  grid (5, 2) x 256.
// ---------------------------------------------------------------------------
__global__ void k_csq(const float* __restrict__ cent, float* __restrict__ csq) {
  __shared__ float red[256];
  const int k = blockIdx.x, side = blockIdx.y, t = threadIdx.x;
  const float* c = cent + ((size_t)side * 16 + k) * C_DIM;
  float s = 0.f;
  for (int i = t; i < C_DIM; i += 256) { float v = c[i]; s += v * v; }
  red[t] = s; __syncthreads();
  for (int st = 128; st > 0; st >>= 1) { if (t < st) red[t] += red[t + st]; __syncthreads(); }
  if (t == 0) csq[side * 8 + k] = red[0];
}

// ---------------------------------------------------------------------------
// 5. kmeans dot maps: g[side][k][img][48x48] = cent_k . feat  (WMMA).
//    grid (144 tiles, 3 imgs, 2 sides) x 32.
// ---------------------------------------------------------------------------
__global__ void k_km_dots(const float* __restrict__ uimgs,
                          const float* __restrict__ cent, float* __restrict__ g) {
  const int tile = blockIdx.x, img = blockIdx.y, side = blockIdx.z;
  const float* feat = uimgs + (size_t)img * IMG;
  v8f acc = {};
  float dummy = 0.f;
  wmma_pdot<false>(cent + (size_t)side * 16 * C_DIM, feat, tile * 16, acc, dummy);
  const int lane = threadIdx.x & 31;
  if (lane < 16) {
    const int pix = tile * 16 + lane;
#pragma unroll
    for (int r = 0; r < 5; ++r)
      g[(((size_t)side * 5 + r) * 3 + img) * HW + pix] = acc[r];
  }
}

__global__ void k_zero(float* __restrict__ p, int n) {
  const int i = blockIdx.x * 256 + threadIdx.x;
  if (i < n) p[i] = 0.f;
}

// ---------------------------------------------------------------------------
// 6. Assignment at 96x96: argmin_k(||c||^2 - 2 g_interp) (||p||^2 cancels),
//    scatter bilinear coefficients into 48x48 per-cluster maps.
// ---------------------------------------------------------------------------
__global__ void k_assign(const float* __restrict__ g, const float* __restrict__ csq,
                         const float* __restrict__ hard,
                         float* __restrict__ coeff, float* __restrict__ den) {
  const int i = blockIdx.x * 256 + threadIdx.x;  // over 2*3*9216
  if (i >= 2 * 3 * 9216) return;
  const int side = i / (3 * 9216);
  const int r2 = i % (3 * 9216);
  const int img = r2 / 9216, p = r2 % 9216;
  const int Y = p / 96, X = p % 96;
  const float w = hard[(((size_t)(img + 1) * 2) + side) * HW + (Y >> 1) * 48 + (X >> 1)];
  if (w == 0.f) return;
  const float ys = Y * 47.f / 95.f, xs = X * 47.f / 95.f;
  const int y0 = (int)floorf(ys); const int y1 = (y0 + 1 < 47) ? y0 + 1 : 47;
  const int x0 = (int)floorf(xs); const int x1 = (x0 + 1 < 47) ? x0 + 1 : 47;
  const float wy = ys - (float)y0, wx = xs - (float)x0;
  const float* gb = g + (size_t)side * 5 * 3 * HW + (size_t)img * HW;
  int best = 0; float bestd = 3.4e38f;
#pragma unroll
  for (int k = 0; k < 5; ++k) {
    const float* gk = gb + (size_t)k * 3 * HW;
    const float t0 = gk[y0 * 48 + x0] * (1.f - wy) + gk[y1 * 48 + x0] * wy;
    const float t1 = gk[y0 * 48 + x1] * (1.f - wy) + gk[y1 * 48 + x1] * wy;
    const float d = csq[side * 8 + k] - 2.f * (t0 * (1.f - wx) + t1 * wx);
    if (d < bestd) { bestd = d; best = k; }   // first-index tie break, like argmin
  }
  float* cc = coeff + ((size_t)side * 16 + best) * KPIX + (size_t)img * HW;
  atomicAdd(&cc[y0 * 48 + x0], w * (1.f - wy) * (1.f - wx));
  atomicAdd(&cc[y0 * 48 + x1], w * (1.f - wy) * wx);
  atomicAdd(&cc[y1 * 48 + x0], w * wy * (1.f - wx));
  atomicAdd(&cc[y1 * 48 + x1], w * wy * wx);
  atomicAdd(&den[side * 8 + best], w);
}

// ---------------------------------------------------------------------------
// 7. Centroid update (WMMA + TDM): num = coeff(16xKPIX) * feat^T(KPIXx2048).
//    Block = 4 waves x 16 channels; wave 0 DMAs shared 16xKCH A-tiles into
//    LDS (tensor_load_to_lds), double-buffered on TENSORcnt; all waves
//    consume A from LDS (ds_load_b128) against contiguous b128 B loads.
//    grid (32 ch-groups, 2 sides) x 128.  2304 % 32 == 0 -> chunks never
//    cross image boundaries.
// ---------------------------------------------------------------------------
__global__ void k_km_update(const float* __restrict__ uimgs,
                            const float* __restrict__ coeff,
                            const float* __restrict__ den,
                            float* __restrict__ cent) {
  __shared__ float abuf[2][16][KCH];
  const int side = blockIdx.y;
  const int wid  = threadIdx.x >> 5;           // wave in block: 0..3
  const int lane = threadIdx.x & 31;
  const int half = lane >> 4, col = lane & 15;
  const int ch0  = (blockIdx.x * 4 + wid) * 16;
  const float* __restrict__ Abase = coeff + (size_t)side * 16 * KPIX;
  const int NCHUNK = KPIX / KCH;               // 27

  if (wid == 0)
    tdm_load_tile(Abase, (unsigned)(uintptr_t)&abuf[0][0][0]);

  v8f acc = {};
  for (int c = 0; c < NCHUNK; ++c) {
    if (wid == 0) __builtin_amdgcn_s_wait_tensorcnt(0);   // chunk c landed
    __syncthreads();                                      // visible to all waves
    if (wid == 0 && c + 1 < NCHUNK)                       // overlap DMA c+1
      tdm_load_tile(Abase + (size_t)(c + 1) * KCH,
                    (unsigned)(uintptr_t)&abuf[(c + 1) & 1][0][0]);

    const float* __restrict__ arow = &abuf[c & 1][col][half * 16];
    const int kg0 = c * KCH;
    for (int kk = 0; kk < KCH; kk += 32) {
      const int kp = kg0 + kk + half * 16;
      const int img = kp / HW, p = kp % HW;
      const float4* __restrict__ f4 =
          (const float4*)(uimgs + (size_t)img * IMG + (size_t)(ch0 + col) * HW + p);
      v16bf a, b;
#pragma unroll
      for (int j4 = 0; j4 < 4; ++j4) {
        const float4 av = *(const float4*)(arow + kk + j4 * 4);  // ds_load_b128
        const float4 bv = f4[j4];                                // global b128
        a[j4 * 4 + 0] = (__bf16)av.x; a[j4 * 4 + 1] = (__bf16)av.y;
        a[j4 * 4 + 2] = (__bf16)av.z; a[j4 * 4 + 3] = (__bf16)av.w;
        b[j4 * 4 + 0] = (__bf16)bv.x; b[j4 * 4 + 1] = (__bf16)bv.y;
        b[j4 * 4 + 2] = (__bf16)bv.z; b[j4 * 4 + 3] = (__bf16)bv.w;
      }
      acc = __builtin_amdgcn_wmma_f32_16x16x32_bf16(false, a, false, b,
                                                    (short)0, acc, false, false);
    }
  }

  if (lane < 16) {
#pragma unroll
    for (int r = 0; r < 5; ++r) {
      const float d = den[side * 8 + r];
      const size_t ci = ((size_t)side * 16 + r) * C_DIM + ch0 + col;
      const float old = cent[ci];
      cent[ci] = (d > 0.f) ? (acc[r] / (d + EPS_F)) : old;
    }
  }
}

// ---------------------------------------------------------------------------
// 8. Build normalized final proto matrix: rows 0-5 = bg set, 6-11 = fg set.
// ---------------------------------------------------------------------------
__global__ void k_final_protos(const float* __restrict__ sp, const float* __restrict__ cent,
                               float* __restrict__ finalP, int epi) {
  __shared__ float red[256];
  const int row = blockIdx.x, t = threadIdx.x;
  const float* src = nullptr;
  if (row == 0)       src = sp + (size_t)epi * 2 * C_DIM;
  else if (row <= 5)  src = cent + (size_t)(row - 1) * C_DIM;            // bg_cls
  else if (row == 6)  src = sp + ((size_t)epi * 2 + 1) * C_DIM;
  else if (row <= 11) src = cent + (size_t)(16 + row - 7) * C_DIM;       // fg_cls
  float s = 0.f;
  if (src) for (int c = t; c < C_DIM; c += 256) { float v = src[c]; s += v * v; }
  red[t] = s; __syncthreads();
  for (int st = 128; st > 0; st >>= 1) { if (t < st) red[t] += red[t + st]; __syncthreads(); }
  const float inv = src ? 1.f / fmaxf(sqrtf(red[0]), EPS_F) : 0.f;
  for (int c = t; c < C_DIM; c += 256)
    finalP[(size_t)row * C_DIM + c] = src ? src[c] * inv : 0.f;
}

// ---------------------------------------------------------------------------
// 9. Final cal_dist (12 protos, WMMA) + per-pixel 2-way softmax -> prob maps.
// ---------------------------------------------------------------------------
__global__ void k_final_caldist(const float* __restrict__ qimg,
                                const float* __restrict__ finalP,
                                const float* __restrict__ invnorm,
                                float* __restrict__ prob, int epi) {
  const int tile = blockIdx.x;
  v8f acc = {};
  float dummy = 0.f;
  wmma_pdot<false>(finalP, qimg, tile * 16, acc, dummy);
  const int lane = threadIdx.x & 31;
  const int half = lane >> 4, col = lane & 15;
  float mb = -3.4e38f, mf = -3.4e38f;
#pragma unroll
  for (int r = 0; r < 8; ++r) {
    const int M = r + half * 8;
    const float v = acc[r];
    if (M <= 5) mb = fmaxf(mb, v);
    else if (M <= 11) mf = fmaxf(mf, v);
  }
  mb = fmaxf(mb, __shfl_xor(mb, 16));
  mf = fmaxf(mf, __shfl_xor(mf, 16));
  if (lane < 16) {
    const int pix = tile * 16 + col;
    const float inv = invnorm[pix];           // image 0 norms from init pass
    const float sb = SCALER_F * inv * mb, sf = SCALER_F * inv * mf;
    const float pb = 1.f / (1.f + __expf(sf - sb));
    prob[((size_t)epi * 2 + 0) * HW + pix] = pb;
    prob[((size_t)epi * 2 + 1) * HW + pix] = 1.f - pb;
  }
}

// ---------------------------------------------------------------------------
// 10. Bilinear 48 -> 417 on (4,2) prob maps.
// ---------------------------------------------------------------------------
__global__ void k_resize_out(const float* __restrict__ prob, float* __restrict__ out) {
  const int i = blockIdx.x * 256 + threadIdx.x;
  if (i >= 4 * 2 * 417 * 417) return;
  const int X = i % 417;
  const int r = i / 417;
  const int Y = r % 417;
  const int ec = r / 417;
  const float ys = Y * 47.f / 416.f, xs = X * 47.f / 416.f;
  const int y0 = (int)floorf(ys); const int y1 = (y0 + 1 < 47) ? y0 + 1 : 47;
  const int x0 = (int)floorf(xs); const int x1 = (x0 + 1 < 47) ? x0 + 1 : 47;
  const float wy = ys - (float)y0, wx = xs - (float)x0;
  const float* pm = prob + (size_t)ec * HW;
  const float t0 = pm[y0 * 48 + x0] * (1.f - wy) + pm[y1 * 48 + x0] * wy;
  const float t1 = pm[y0 * 48 + x1] * (1.f - wy) + pm[y1 * 48 + x1] * wy;
  out[i] = t0 * (1.f - wx) + t1 * wx;
}

// ---------------------------------------------------------------------------
extern "C" void kernel_launch(void* const* d_in, const int* in_sizes, int n_in,
                              void* d_out, int out_size, void* d_ws, size_t ws_size,
                              hipStream_t stream) {
  (void)in_sizes; (void)n_in; (void)out_size; (void)ws_size;
  const float* sp = (const float*)d_in[0];   // (4,2,2048)
  const float* qf = (const float*)d_in[1];   // (1,4,2048,48,48)
  const float* un = (const float*)d_in[2];   // (12,2048,48,48)
  float* out = (float*)d_out;
  float* ws  = (float*)d_ws;

  // workspace layout (floats), ~1.9 MB total
  size_t o = 0;
  float* protoN2 = ws + o; o += (size_t)16 * C_DIM;       // 32768
  float* invnorm = ws + o; o += (size_t)4 * HW;           // 9216
  float* hard    = ws + o; o += (size_t)4 * 2 * HW;       // 18432
  float* cent    = ws + o; o += (size_t)2 * 16 * C_DIM;   // 65536
  float* csq     = ws + o; o += 16;
  float* g       = ws + o; o += (size_t)2 * 5 * 3 * HW;   // 69120
  float* coeff   = ws + o; o += (size_t)2 * 16 * KPIX;    // 221184
  float* den     = ws + o; o += 16;                       // contiguous after coeff
  float* finalP  = ws + o; o += (size_t)16 * C_DIM;       // 32768
  float* prob    = ws + o; o += (size_t)4 * 2 * HW;       // 18432

  float* masks = out + (size_t)4 * 2 * 417 * 417;

  for (int epi = 0; epi < 4; ++epi) {
    const float* qimg  = qf + (size_t)epi * IMG;
    const float* uimgs = un + (size_t)epi * 3 * IMG;

    k_norm_protos<<<16, 256, 0, stream>>>(sp, protoN2, epi);
    k_init_caldist<<<dim3(144, 4), 32, 0, stream>>>(qimg, uimgs, protoN2,
                                                    invnorm, hard, masks, epi);
    k_cent_init<<<dim3(16, 8), 256, 0, stream>>>(uimgs, cent);
    k_csq<<<dim3(5, 2), 256, 0, stream>>>(cent, csq);

    for (int it = 0; it < 10; ++it) {
      k_km_dots<<<dim3(144, 3, 2), 32, 0, stream>>>(uimgs, cent, g);
      const int nz = 2 * 16 * KPIX + 16;                  // coeff + den
      k_zero<<<(nz + 255) / 256, 256, 0, stream>>>(coeff, nz);
      k_assign<<<(2 * 3 * 9216 + 255) / 256, 256, 0, stream>>>(g, csq, hard,
                                                               coeff, den);
      k_km_update<<<dim3(32, 2), 128, 0, stream>>>(uimgs, coeff, den, cent);
      k_csq<<<dim3(5, 2), 256, 0, stream>>>(cent, csq);
    }

    k_final_protos<<<16, 256, 0, stream>>>(sp, cent, finalP, epi);
    k_final_caldist<<<144, 32, 0, stream>>>(qimg, finalP, invnorm, prob, epi);
  }

  k_resize_out<<<(4 * 2 * 417 * 417 + 255) / 256, 256, 0, stream>>>(prob, out);
}
